// FPSANConv2d_38036230374126
// MI455X (gfx1250) — matrix operand order repaired and checked by reading, not compiled, not run on hardware
//
#include <hip/hip_runtime.h>

typedef __attribute__((ext_vector_type(16))) _Float16 v16h;
typedef __attribute__((ext_vector_type(8)))  _Float16 v8h;
typedef __attribute__((ext_vector_type(8)))  float    v8f;

#define CIN   256
#define COUT  256
#define HH    128
#define WW    128
#define NBATCH 16
#define K9    9
#define NWELT (COUT*CIN*K9)   // 589824

#define MT    128     // co tile per workgroup
#define CB    32      // ci chunk = WMMA K
#define CBP   40      // padded ci stride in LDS: 80B rows -> 2-way instead of 32-way conflicts
#define WPAD  132     // 128 + halo(2) + pad

union U16 { v16h v; v8h h[2]; };

// ---------------- scale = max(mean|w|, 1e-5), deterministic 2-stage ----------------
__global__ void ternary_reduce_partial(const float* __restrict__ w, float* __restrict__ partials) {
    __shared__ float sm[256];
    const int tid = threadIdx.x;
    float s = 0.f;
    for (int i = blockIdx.x * 256 + tid; i < NWELT; i += 512 * 256) s += fabsf(w[i]);
    sm[tid] = s; __syncthreads();
    for (int off = 128; off > 0; off >>= 1) {
        if (tid < off) sm[tid] += sm[tid + off];
        __syncthreads();
    }
    if (tid == 0) partials[blockIdx.x] = sm[0];
}

__global__ void ternary_reduce_final(const float* __restrict__ partials, float* __restrict__ scale) {
    __shared__ float sm[256];
    const int tid = threadIdx.x;
    sm[tid] = partials[tid] + partials[tid + 256];
    __syncthreads();
    for (int off = 128; off > 0; off >>= 1) {
        if (tid < off) sm[tid] += sm[tid + off];
        __syncthreads();
    }
    if (tid == 0) scale[0] = fmaxf(sm[0] / (float)NWELT, 1e-5f);
}

// -------- ternary quantize + re-layout: wq[co][tap][ci] = clip(rint(w/s),-1,1) -----
__global__ void ternary_quantize(const float* __restrict__ w, const float* __restrict__ scale,
                                 _Float16* __restrict__ wq) {
    const int j = blockIdx.x * 256 + threadIdx.x;     // j indexes [co][tap][ci]
    if (j < NWELT) {
        const int ci  = j & 255;
        const int t2  = j >> 8;
        const int tap = t2 % K9;
        const int co  = t2 / K9;
        const float s = scale[0];
        float t = rintf(w[(co * CIN + ci) * K9 + tap] / s);   // round-half-even = jnp.round
        t = fminf(fmaxf(t, -1.f), 1.f);
        wq[j] = (_Float16)t;            // {-1,0,+1} exact in f16; scale applied in epilogue
    }
}

// ---------------- implicit-GEMM conv on WMMA f16 -----------------------------------
// grid = (H, N, COUT/MT); block = 256 (8 waves arranged 4(M)x2(N)).
// Wave (wvM,wvN) owns co [co0+wvM*32,+32) x w [wvN*64,+64): 2x4 = 8 WMMA tiles,
// needing only 2 A + 4 B fragments per tap (12 ds_load_b128 per 8 WMMAs).
__global__ __launch_bounds__(256)
void conv3x3_wmma_f16(const float* __restrict__ x, const _Float16* __restrict__ wq,
                      const float* __restrict__ bias, const float* __restrict__ scale,
                      float* __restrict__ out) {
    __shared__ alignas(16) _Float16 xs[3 * WPAD * CBP];  // [row][wp][ci(pad 40)], wp = w_img+1
    __shared__ alignas(16) _Float16 wl[MT * K9 * CB];    // [co_local][tap][ci]

    const int h    = blockIdx.x;
    const int n    = blockIdx.y;
    const int co0  = blockIdx.z * MT;
    const int tid  = threadIdx.x;
    const int lane = tid & 31;
    const int wv   = tid >> 5;       // wave id 0..7
    const int wvN  = wv & 1;         // N group: w block of 64
    const int wvM  = wv >> 1;        // M group: co block of 32
    const int lm   = lane & 15;
    const int half = lane >> 4;

    const float s = scale[0];

    v8f acc[2][4];
    #pragma unroll
    for (int mi = 0; mi < 2; ++mi)
        #pragma unroll
        for (int nj = 0; nj < 4; ++nj) acc[mi][nj] = (v8f){};

    // per-wave A bases in LDS: rows (wvM*32 + mi*16 + lm), layout [tap][ci]
    const _Float16* apbase[2] = {
        &wl[(wvM * 32 +      lm) * K9 * CB],
        &wl[(wvM * 32 + 16 + lm) * K9 * CB]
    };
    // A fragment (16-bit A 16x32): elems 0..7 -> K=8*half.., elems 8..15 -> K=16+8*half..
    auto loadTapA = [&](int tap, U16 (&A)[2]) {
        #pragma unroll
        for (int mi = 0; mi < 2; ++mi) {
            const _Float16* ap = apbase[mi] + tap * CB;
            A[mi].h[0] = *(const v8h*)(ap + 8 * half);
            A[mi].h[1] = *(const v8h*)(ap + 16 + 8 * half);
        }
    };
    // B fragment (32x16 f16: col=lm, elems 0..7 -> K=8*half.., elems 8..15 -> K=16+8*half..)
    auto loadTapB = [&](int tap, U16 (&B)[4]) {
        const int kh = tap / 3, kw = tap % 3;
        #pragma unroll
        for (int nj = 0; nj < 4; ++nj) {
            const int wp = wvN * 64 + nj * 16 + lm + kw;
            const v8h* bp = (const v8h*)&xs[(kh * WPAD + wp) * CBP + 8 * half];
            B[nj].h[0] = bp[0];
            B[nj].h[1] = bp[2];
        }
    };

    for (int ci0 = 0; ci0 < CIN; ci0 += CB) {
        __syncthreads();
        // ---- stage x rows h-1..h+1 (f32 -> f16), layout [row][wp][ci] ----
        // lanes sweep wp: global b32 loads coalesced; LDS stores 80B apart -> 2-way conflicts
        for (int idx = tid; idx < 3 * WPAD * CB; idx += 256) {
            const int wp  = idx % WPAD;
            const int t2  = idx / WPAD;
            const int row = t2 % 3;
            const int ci  = t2 / 3;
            const int wi  = wp - 1;
            const int hi  = h + row - 1;
            float v = 0.f;
            if ((unsigned)wi < WW && (unsigned)hi < HH)
                v = x[(((n * CIN + ci0 + ci) * HH + hi) * WW) + wi];
            xs[(row * WPAD + wp) * CBP + ci] = (_Float16)v;
        }
        // ---- stage weight chunk with async global->LDS DMA (no VGPR round-trip) ----
        // global wq[co][tap][256ci] chunk -> LDS wl[co][tap][32ci]; 4608 x 16B, tracked
        // by ASYNCcnt. 4608/256 = 18 iters for every thread: uniform, EXEC stays full.
        for (int idx = tid; idx < MT * K9 * 4; idx += 256) {
            const int u   = idx & 3;
            const int ct  = idx >> 2;
            const int tap = ct % K9;
            const int co  = ct / K9;
            const unsigned ldsoff = (unsigned)(size_t)&wl[(co * K9 + tap) * CB + u * 8];
            const unsigned goff   = (unsigned)((((co0 + co) * K9 + tap) * CIN + ci0 + u * 8)
                                               * sizeof(_Float16));
            asm volatile("global_load_async_to_lds_b128 %0, %1, %2 offset:0"
                         :: "v"(ldsoff), "v"(goff), "s"(wq) : "memory");
        }
        // ---- prefetch next chunk's x rows into cache while this chunk computes ----
        if (ci0 + CB < CIN) {
            for (int p = tid; p < 3 * (WW / 16) * CB; p += 256) {   // 64B granules, 3/thread
                const int wseg = p % (WW / 16);
                const int t2   = p / (WW / 16);
                const int row  = t2 % 3;
                const int ci   = t2 / 3;
                const int hi   = h + row - 1;
                if ((unsigned)hi < HH)
                    __builtin_prefetch(&x[(((n * CIN + ci0 + CB + ci) * HH + hi) * WW)
                                          + wseg * 16], 0, 0);
            }
        }
        asm volatile("s_wait_asynccnt 0" ::: "memory");
        __syncthreads();

        // ---- tap-pipelined WMMA: prefetch tap t+1 fragments during tap t burst ----
        U16 a[2][2], b[2][4];
        loadTapA(0, a[0]);
        loadTapB(0, b[0]);
        #pragma unroll
        for (int tap = 0; tap < 9; ++tap) {
            const int cur = tap & 1;
            const int nxt = cur ^ 1;
            if (tap < 8) {
                loadTapA(tap + 1, a[nxt]);
                loadTapB(tap + 1, b[nxt]);
            }
            #pragma unroll
            for (int mi = 0; mi < 2; ++mi)
                #pragma unroll
                for (int nj = 0; nj < 4; ++nj)
                    acc[mi][nj] = __builtin_amdgcn_wmma_f32_16x16x32_f16(
                        false, a[cur][mi].v, false, b[cur][nj].v, (short)0,
                        acc[mi][nj], false, false);
        }
    }

    // ---- epilogue: D layout (M = r + 8*half, N = lm) -> out; apply scale + bias ----
    #pragma unroll
    for (int mi = 0; mi < 2; ++mi) {
        float bv[8];
        #pragma unroll
        for (int r = 0; r < 8; ++r)
            bv[r] = bias[co0 + wvM * 32 + mi * 16 + 8 * half + r];
        #pragma unroll
        for (int nj = 0; nj < 4; ++nj) {
            const int wcol = wvN * 64 + nj * 16 + lm;
            #pragma unroll
            for (int r = 0; r < 8; ++r) {
                const int co = co0 + wvM * 32 + mi * 16 + 8 * half + r;
                out[(((n * COUT + co) * HH + h) * WW) + wcol] = acc[mi][nj][r] * s + bv[r];
            }
        }
    }
}

extern "C" void kernel_launch(void* const* d_in, const int* in_sizes, int n_in,
                              void* d_out, int out_size, void* d_ws, size_t ws_size,
                              hipStream_t stream) {
    const float* x    = (const float*)d_in[0];
    const float* wt   = (const float*)d_in[1];
    const float* bias = (const float*)d_in[2];
    float* out        = (float*)d_out;

    float*    partials = (float*)d_ws;                        // 512 floats
    float*    scale    = (float*)((char*)d_ws + 2048);        // 1 float
    _Float16* wq       = (_Float16*)((char*)d_ws + 4096);     // 589824 f16 (~1.18 MB)

    ternary_reduce_partial<<<512, 256, 0, stream>>>(wt, partials);
    ternary_reduce_final<<<1, 256, 0, stream>>>(partials, scale);
    ternary_quantize<<<(NWELT + 255) / 256, 256, 0, stream>>>(wt, scale, wq);

    dim3 grid(HH, NBATCH, COUT / MT);   // 128 x 16 x 2
    conv3x3_wmma_f16<<<grid, 256, 0, stream>>>(x, wq, bias, scale, out);
}
